// CovLayer_65377992179732
// MI455X (gfx1250) — matrix operand order repaired and checked by reading, not compiled, not run on hardware
//
#include <hip/hip_runtime.h>

// ---------------------------------------------------------------------------
// CovLayer fused kernel for MI455X (gfx1250, wave32, WMMA)
//   out[i,j,:] = LN( sum_s (xl[s,i,:]*w[s]) outer xr[s,j,:] ) @ W + b
// Shapes fixed from reference: S=128, N=256, E=32, D2=1024, OUT=64.
// ---------------------------------------------------------------------------

typedef _Float16 f16;
typedef __attribute__((ext_vector_type(8)))  _Float16 v8h;
typedef __attribute__((ext_vector_type(16))) _Float16 v16h;
typedef __attribute__((ext_vector_type(8)))  float    v8f;

#define S_DIM   128
#define N_RES   256
#define E_DIM   32
#define D2      1024
#define OUT_DIM 64
#define TI      4          // i-residues per workgroup
#define TJ      4          // j-residues per workgroup
#define ROWPAD  8          // f16 row padding -> bank-conflict mitigation
#define XROW    (S_DIM + ROWPAD)   // 136 halves per [e] row
#define PROW    (D2 + ROWPAD)      // 1032 halves per pair row

// floats: gamma(1024) + beta(1024) + bias(64) ; halves: xl + xr + P
#define SMEM_FLOATS (1024 + 1024 + 64)
#define SMEM_BYTES  (SMEM_FLOATS*4 + (TI*E_DIM*XROW)*2 + (TJ*E_DIM*XROW)*2 + (16*PROW)*2)

static __device__ __forceinline__ v16h ld2(const f16* p0, const f16* p1) {
    union { v16h v; v8h h[2]; } u;
    u.h[0] = *(const v8h*)p0;   // 16B-aligned b128
    u.h[1] = *(const v8h*)p1;
    return u.v;
}

static __device__ __forceinline__ v8f wmma_f16(v16h a, v16h b, v8f c) {
    return __builtin_amdgcn_wmma_f32_16x16x32_f16(
        /*neg_a=*/false, a, /*neg_b=*/false, b,
        /*c_mod=*/(short)0, c, /*reuse_a=*/false, /*reuse_b=*/false);
}

// ---------------------------------------------------------------------------
// Prep: Wt[o][k] = (f16) W[k][o]   (64 x 1024 f16, contiguous in k)
// ---------------------------------------------------------------------------
__global__ void prep_wt_kernel(const float* __restrict__ W, f16* __restrict__ Wt) {
    int idx = blockIdx.x * blockDim.x + threadIdx.x;   // 0 .. 65535
    int o = idx >> 10;
    int k = idx & (D2 - 1);
    Wt[idx] = (f16)W[k * OUT_DIM + o];
}

// ---------------------------------------------------------------------------
// Main fused kernel
// ---------------------------------------------------------------------------
__global__ __launch_bounds__(256)
void cov_main_kernel(const float* __restrict__ xl, const float* __restrict__ xr,
                     const float* __restrict__ sw, const float* __restrict__ gamma,
                     const float* __restrict__ beta, const f16* __restrict__ Wt,
                     const float* __restrict__ bias, float* __restrict__ out)
{
    extern __shared__ char smem_raw[];
    float* s_gamma = (float*)smem_raw;          // 1024
    float* s_beta  = s_gamma + 1024;            // 1024
    float* s_bias  = s_beta + 1024;             // 64
    f16*   s_xl    = (f16*)(s_bias + 64);       // [TI][32][XROW]  (e-major, s inner)
    f16*   s_xr    = s_xl + TI*E_DIM*XROW;      // [TJ][32][XROW]
    f16*   s_p     = s_xr + TJ*E_DIM*XROW;      // [16][PROW] normalized pairs (f16)
    float* s_part  = (float*)s_xl;              // overlay: projection partials (4 KB),
                                                // s_xl is dead after the op-GEMM phase

    const int tid = threadIdx.x;
    const int i0  = blockIdx.x * TI;
    const int j0  = blockIdx.y * TJ;

    // ---- stage gamma/beta/bias ----
    for (int d = tid; d < D2; d += 256) { s_gamma[d] = gamma[d]; s_beta[d] = beta[d]; }
    if (tid < OUT_DIM) s_bias[tid] = bias[tid];

    // ---- stage xl (seq-weighted) and xr tiles, transposed [e][s], f16 ----
    // Each thread handles 2 consecutive s for one e -> packed b32 LDS store.
    for (int idx = tid; idx < TI * (S_DIM / 2) * E_DIM; idx += 256) {
        int ii = idx >> 11;              // / (64*32)
        int r  = idx & 2047;
        int s  = (r >> 5) << 1;          // even s
        int e  = r & 31;
        float w0 = sw[s], w1 = sw[s + 1];
        const float* pl = xl + (s * N_RES + (i0 + ii)) * E_DIM + e;
        const float* pr = xr + (s * N_RES + (j0 + ii)) * E_DIM + e;
        union { f16 h[2]; unsigned u; } pkl, pkr;
        pkl.h[0] = (f16)(pl[0] * w0);
        pkl.h[1] = (f16)(pl[N_RES * E_DIM] * w1);
        pkr.h[0] = (f16)(pr[0]);
        pkr.h[1] = (f16)(pr[N_RES * E_DIM]);
        *(unsigned*)&s_xl[(ii * E_DIM + e) * XROW + s] = pkl.u;
        *(unsigned*)&s_xr[(ii * E_DIM + e) * XROW + s] = pkr.u;
    }
    __syncthreads();

    const int wave  = tid >> 5;      // 0..7
    const int lane  = tid & 31;
    const int laneN = lane & 15;
    const int hi    = lane >> 4;     // half-wave select

    // ---- per-pair 32x32 outer-product GEMM + LayerNorm (2 pairs / wave) ----
    for (int pp = 0; pp < 2; ++pp) {
        const int p  = wave * 2 + pp;       // pair id 0..15
        const int ii = p >> 2;
        const int jj = p & 3;

        v8f c00 = {}; v8f c01 = {}; v8f c10 = {}; v8f c11 = {};
        const f16* xlb = s_xl + ii * E_DIM * XROW;
        const f16* xrb = s_xr + jj * E_DIM * XROW;

        #pragma unroll
        for (int kc = 0; kc < 4; ++kc) {
            const int sA = kc * 32 + hi * 8;    // A: K slots {0..7,16..23} / {8..15,24..31}
            const int sB = kc * 32 + hi * 8;    // B: K slots {0..7,16..23} / {8..15,24..31}
            const f16* a0p = xlb + (      laneN) * XROW + sA;
            const f16* a1p = xlb + (16  + laneN) * XROW + sA;
            const f16* b0p = xrb + (      laneN) * XROW + sB;
            const f16* b1p = xrb + (16  + laneN) * XROW + sB;
            v16h a0 = ld2(a0p, a0p + 16);
            v16h a1 = ld2(a1p, a1p + 16);
            v16h b0 = ld2(b0p, b0p + 16);
            v16h b1 = ld2(b1p, b1p + 16);
            c00 = wmma_f16(a0, b0, c00);
            c01 = wmma_f16(a0, b1, c01);
            c10 = wmma_f16(a1, b0, c10);
            c11 = wmma_f16(a1, b1, c11);
        }

        // ---- LayerNorm over the 1024 values held by this wave ----
        float sum = 0.f, sq = 0.f;
        #pragma unroll
        for (int g = 0; g < 8; ++g) {
            float v0 = c00[g], v1 = c01[g], v2 = c10[g], v3 = c11[g];
            sum += (v0 + v1) + (v2 + v3);
            sq  += (v0*v0 + v1*v1) + (v2*v2 + v3*v3);
        }
        #pragma unroll
        for (int m = 16; m >= 1; m >>= 1) {
            sum += __shfl_xor(sum, m, 32);
            sq  += __shfl_xor(sq,  m, 32);
        }
        const float mu   = sum * (1.f / 1024.f);
        const float var  = sq * (1.f / 1024.f) - mu * mu;
        const float rstd = rsqrtf(var + 1e-5f);

        f16* prow = s_p + p * PROW;
        #pragma unroll
        for (int g = 0; g < 8; ++g) {
            const int lA = g + hi * 8;          // C/D layout: M = g (+8 for hi lanes)
            int d0 = (     lA) * 32 +      laneN;
            int d1 = (     lA) * 32 + 16 + laneN;
            int d2 = (16 + lA) * 32 +      laneN;
            int d3 = (16 + lA) * 32 + 16 + laneN;
            prow[d0] = (f16)((c00[g] - mu) * rstd * s_gamma[d0] + s_beta[d0]);
            prow[d1] = (f16)((c01[g] - mu) * rstd * s_gamma[d1] + s_beta[d1]);
            prow[d2] = (f16)((c10[g] - mu) * rstd * s_gamma[d2] + s_beta[d2]);
            prow[d3] = (f16)((c11[g] - mu) * rstd * s_gamma[d3] + s_beta[d3]);
        }
    }
    __syncthreads();   // s_p complete; s_xl region now reusable as s_part

    // ---- projection: [16 pairs x 1024] @ Wt^T -> [16 x 64] ----
    // All 8 waves: wave = {kh,tj}; each computes half of K, partials combined in LDS.
    {
        const int tj = wave & 3;                  // 16-wide output column tile
        const int kh = wave >> 2;                 // K half: 0 -> [0,512), 1 -> [512,1024)
        v8f acc = {};
        const f16* wrow = Wt + (tj * 16 + laneN) * D2;   // Wt[o][k], contiguous k
        const f16* arow = s_p + laneN * PROW;            // pair row (A: M = laneN)
        __builtin_prefetch(wrow + kh * 512, 0, 1);       // global_prefetch_b8
        #pragma unroll 4
        for (int kc = kh * 16; kc < kh * 16 + 16; ++kc) {
            const int kA = kc * 32 + hi * 8;
            const int kB = kc * 32 + hi * 8;
            v16h a = ld2(arow + kA, arow + kA + 16);
            v16h b = ld2(wrow + kB, wrow + kB + 16);
            acc = wmma_f16(a, b, acc);
        }

        if (kh == 1) {                            // stash upper-K partial
            float* sp = s_part + (tj * 32 + lane) * 8;
            #pragma unroll
            for (int g = 0; g < 8; ++g) sp[g] = acc[g];
        }
        __syncthreads();
        if (kh == 0) {                            // combine + epilogue
            const float* sp = s_part + (tj * 32 + lane) * 8;
            const int o  = tj * 16 + laneN;
            const float bo = s_bias[o];
            #pragma unroll
            for (int g = 0; g < 8; ++g) {
                const int p = g + hi * 8;         // D layout: M = pair index
                const int i = i0 + (p >> 2);
                const int j = j0 + (p & 3);
                out[((i * N_RES + j) * OUT_DIM) + o] = acc[g] + sp[g] + bo;
            }
        }
    }
}

// ---------------------------------------------------------------------------
extern "C" void kernel_launch(void* const* d_in, const int* in_sizes, int n_in,
                              void* d_out, int out_size, void* d_ws, size_t ws_size,
                              hipStream_t stream) {
    const float* xl    = (const float*)d_in[0];   // [1,128,256,32]
    const float* xr    = (const float*)d_in[1];   // [1,128,256,32]
    const float* sw    = (const float*)d_in[2];   // [1,128,1,1]
    const float* gamma = (const float*)d_in[3];   // [1024]
    const float* beta  = (const float*)d_in[4];   // [1024]
    const float* W     = (const float*)d_in[5];   // [1024,64]
    const float* bias  = (const float*)d_in[6];   // [64]
    float*       out   = (float*)d_out;           // [1,256,256,64]
    f16*         Wt    = (f16*)d_ws;              // 64*1024 f16 = 128 KB scratch

    hipFuncSetAttribute((const void*)cov_main_kernel,
                        hipFuncAttributeMaxDynamicSharedMemorySize, (int)SMEM_BYTES);

    prep_wt_kernel<<<dim3((D2 * OUT_DIM) / 256), dim3(256), 0, stream>>>(W, Wt);

    dim3 grid(N_RES / TI, N_RES / TJ);   // 64 x 64 workgroups
    cov_main_kernel<<<grid, dim3(256), SMEM_BYTES, stream>>>(
        xl, xr, sw, gamma, beta, Wt, bias, out);
}
